// MultiHead_TADA_68822555951555
// MI455X (gfx1250) — compile-verified, matrix operand order checked by reading, not run on hardware
//
#include <hip/hip_runtime.h>
#include <hip/hip_bf16.h>

// ---------------------------------------------------------------------------
// Differential attention (DiffAttn), B=4, N=1024, DIM=1024, H=16, HD=64.
// Compute-bound (~86 GFLOP vs ~50MB HBM traffic) -> everything through
// v_wmma_f32_16x16x32_bf16. bf16 workspace between stages keeps intermediate
// traffic (~100MB) L2-resident (192MB L2).
//  - SCALE (=0.125, a power of two) folded into the Q-projection epilogue:
//    removes post-WMMA VALU scaling + hazard NOPs in the attention loop.
//  - GEMMs double-buffered: global loads for tile t+1 in flight under the
//    WMMAs of tile t.
//  - Attention K/V/Q staging uses GLOBAL_LOAD_ASYNC_TO_LDS_B128 (ASYNCcnt)
//    when the builtin is available.
// ---------------------------------------------------------------------------

#define DIM_   1024
#define H_     16
#define HD_    64
#define B_     4
#define NSEQ_  1024
#define SCALE_ 0.125f        // 64^-0.5 (exact power of two)
#define LAMBDA_INIT_ 0.1f

typedef __attribute__((ext_vector_type(16))) __bf16 v16bf;
typedef __attribute__((ext_vector_type(8)))  float  v8f;
typedef int v4i_ __attribute__((vector_size(4 * sizeof(int))));   // builtin's pointee type

#if __has_builtin(__builtin_amdgcn_global_load_async_to_lds_b128)
#define HAVE_ASYNC_LDS 1
#else
#define HAVE_ASYNC_LDS 0
#endif

#if HAVE_ASYNC_LDS
__device__ __forceinline__ void async_cp16(const unsigned short* g, unsigned short* l) {
    // 16B global -> LDS, tracked by ASYNCcnt (no VGPR data movement)
    __builtin_amdgcn_global_load_async_to_lds_b128((v4i_*)g, (v4i_*)l, 0, 0);
}
__device__ __forceinline__ void async_wait0() {
#if __has_builtin(__builtin_amdgcn_s_wait_asynccnt)
    __builtin_amdgcn_s_wait_asynccnt(0);
#else
    asm volatile("s_wait_asynccnt 0" ::: "memory");
#endif
}
#endif

__device__ __forceinline__ unsigned short f2bf(float f) {
    unsigned int u = __builtin_bit_cast(unsigned int, f);
    unsigned int lsb = (u >> 16) & 1u;
    u += 0x7FFFu + lsb;                       // round to nearest even
    return (unsigned short)(u >> 16);
}

__device__ __forceinline__ v8f zero8() {
    v8f z;
#pragma unroll
    for (int j = 0; j < 8; ++j) z[j] = 0.f;
    return z;
}

__device__ __forceinline__ v8f wmma_bf16(v16bf a, v16bf b, v8f c) {
    // D = A(16x32 bf16) x B(32x16 bf16) + C(16x16 f32)
    return __builtin_amdgcn_wmma_f32_16x16x32_bf16(
        false, a, false, b, (short)0, c, false, false);
}

// A-fragment (16x32 bf16): lane L<16 holds row M=L; lane halves split K.
//   VGPR0-3 : K = kb..kb+7   (kb = (lane>>4)*8)
//   VGPR4-7 : K = 16+kb..16+kb+7
// `base` is a row-major [rows][ldm] bf16(ushort) tile. The same pattern reads
// a column-major B (i.e. B = Ktile^T).
__device__ __forceinline__ v16bf frag_rowmajor(const unsigned short* base, int ldm,
                                               int row0, int kbase0) {
    int lane = threadIdx.x & 31;
    const unsigned short* p = base + (size_t)(row0 + (lane & 15)) * ldm + kbase0;
    int kb = (lane >> 4) * 8;
    v16bf a;
#pragma unroll
    for (int j = 0; j < 8; ++j) {
        a[j]     = __builtin_bit_cast(__bf16, p[kb + j]);
        a[j + 8] = __builtin_bit_cast(__bf16, p[16 + kb + j]);
    }
    return a;
}

// B-fragment from a row-major KxN tile (lane = column, rows strided).
__device__ __forceinline__ v16bf frag_kn(const unsigned short* base, int ldn,
                                         int col0, int kbase0) {
    int lane = threadIdx.x & 31;
    int col  = col0 + (lane & 15);
    int kb   = kbase0 + (lane >> 4) * 8;
    v16bf b;
#pragma unroll
    for (int j = 0; j < 8; ++j) {
        b[j]     = __builtin_bit_cast(__bf16, base[(size_t)(kb + j) * ldn + col]);
        b[j + 8] = __builtin_bit_cast(__bf16, base[(size_t)(kb + 16 + j) * ldn + col]);
    }
    return b;
}

// ---------------------------------------------------------------------------
// lambda[h] = exp(<lq1,lk1>) - exp(<lq2,lk2>) + LAMBDA_INIT
// ---------------------------------------------------------------------------
__global__ void lambda_kernel(const float* __restrict__ lq1, const float* __restrict__ lk1,
                              const float* __restrict__ lq2, const float* __restrict__ lk2,
                              float* __restrict__ lam) {
    int h = blockIdx.x;
    int lane = threadIdx.x;               // 32 threads (one wave32)
    float s1 = 0.f, s2 = 0.f;
    for (int d = lane; d < HD_; d += 32) {
        s1 += lq1[h * HD_ + d] * lk1[h * HD_ + d];
        s2 += lq2[h * HD_ + d] * lk2[h * HD_ + d];
    }
#pragma unroll
    for (int off = 16; off >= 1; off >>= 1) {
        s1 += __shfl_xor(s1, off, 32);
        s2 += __shfl_xor(s2, off, 32);
    }
    if (lane == 0) lam[h] = __expf(s1) - __expf(s2) + LAMBDA_INIT_;
}

// ---------------------------------------------------------------------------
// Double-buffered bf16 WMMA GEMM: C[MxN] = (A[MxK] * W[KxN]) * cscale
//   A_F32:   A is fp32 (converted to bf16 while staging) else bf16 (ushort)
//   OUT_F32: C is fp32 else bf16 (ushort)
// Block: 256 threads (8 waves), BM=128 BN=128 BK=32, 2x buffered LDS (32KB).
// Wave (wm in {0,64}, wn in {0..96}) owns a 64x32 region = 4x2 WMMA tiles.
// Pipeline: fetch(t+1) -> WMMA(t) -> stage(t+1) -> barrier.
// ---------------------------------------------------------------------------
template <bool A_F32, bool OUT_F32>
__global__ __launch_bounds__(256) void gemm_wmma(const void* __restrict__ Aptr,
                                                 const float* __restrict__ W,
                                                 void* __restrict__ Cptr,
                                                 int M, int Kd, int N, float cscale) {
    constexpr int BM = 128, BN = 128, BK = 32;
    __shared__ unsigned short As[2][BM * BK];   // 2 x 8 KB
    __shared__ unsigned short Bs[2][BK * BN];   // 2 x 8 KB

    const int tid  = threadIdx.x;
    const int bm   = blockIdx.y * BM;
    const int bn   = blockIdx.x * BN;
    const int wave = tid >> 5;
    const int wm   = (wave >> 2) * 64;
    const int wn   = (wave & 3) * 32;

    const float* Af = (const float*)Aptr;
    const unsigned short* Ah = (const unsigned short*)Aptr;

    float4 ar[4];   // A tile staging (fp32 path)
    uint4  arh[2];  // A tile staging (bf16 path)
    float4 wr[4];   // W tile staging

    auto fetch = [&](int k0) {
#pragma unroll
        for (int it = 0; it < 4; ++it) {
            int idx = (tid + it * 256) * 4;
            if constexpr (A_F32) {
                int r = idx >> 5, c = idx & 31;
                ar[it] = *(const float4*)(Af + (size_t)(bm + r) * Kd + k0 + c);
            }
            int rb = idx >> 7, cb = idx & 127;
            wr[it] = *(const float4*)(W + (size_t)(k0 + rb) * N + bn + cb);
            if (k0 + BK < Kd)   // prefetch tile after next -> global_prefetch_b8
                __builtin_prefetch(W + (size_t)(k0 + BK + rb) * N + bn + cb, 0, 0);
        }
        if constexpr (!A_F32) {
#pragma unroll
            for (int it = 0; it < 2; ++it) {
                int idx = (tid + it * 256) * 8;
                int r = idx >> 5, c = idx & 31;
                arh[it] = *(const uint4*)(Ah + (size_t)(bm + r) * Kd + k0 + c);
            }
        }
    };
    auto stage = [&](int buf) {
#pragma unroll
        for (int it = 0; it < 4; ++it) {
            int idx = (tid + it * 256) * 4;
            if constexpr (A_F32) {
                int r = idx >> 5, c = idx & 31;
                unsigned short* d = &As[buf][r * BK + c];
                d[0] = f2bf(ar[it].x); d[1] = f2bf(ar[it].y);
                d[2] = f2bf(ar[it].z); d[3] = f2bf(ar[it].w);
            }
            int rb = idx >> 7, cb = idx & 127;
            unsigned short* d = &Bs[buf][rb * BN + cb];
            d[0] = f2bf(wr[it].x); d[1] = f2bf(wr[it].y);
            d[2] = f2bf(wr[it].z); d[3] = f2bf(wr[it].w);
        }
        if constexpr (!A_F32) {
#pragma unroll
            for (int it = 0; it < 2; ++it) {
                int idx = (tid + it * 256) * 8;
                int r = idx >> 5, c = idx & 31;
                *(uint4*)&As[buf][r * BK + c] = arh[it];
            }
        }
    };

    v8f acc[4][2];
#pragma unroll
    for (int mi = 0; mi < 4; ++mi)
#pragma unroll
        for (int ni = 0; ni < 2; ++ni) acc[mi][ni] = zero8();

    fetch(0);
    stage(0);
    __syncthreads();

    for (int k0 = 0; k0 < Kd; k0 += BK) {
        const int  cur  = (k0 >> 5) & 1;
        const bool more = (k0 + BK) < Kd;
        if (more) fetch(k0 + BK);        // global loads fly under the WMMAs

        v16bf af[4], bfr[2];
#pragma unroll
        for (int mi = 0; mi < 4; ++mi)
            af[mi] = frag_rowmajor(As[cur], BK, wm + mi * 16, 0);
#pragma unroll
        for (int ni = 0; ni < 2; ++ni)
            bfr[ni] = frag_kn(Bs[cur], BN, wn + ni * 16, 0);
#pragma unroll
        for (int mi = 0; mi < 4; ++mi)
#pragma unroll
            for (int ni = 0; ni < 2; ++ni)
                acc[mi][ni] = wmma_bf16(af[mi], bfr[ni], acc[mi][ni]);

        if (more) stage(1 - cur);        // convert + ds_store into other buffer
        __syncthreads();
    }

    // ---- epilogue: C/D layout -> global (scaled) ----
    const int lane = tid & 31;
    const int ln = lane & 15, hh = lane >> 4;
#pragma unroll
    for (int mi = 0; mi < 4; ++mi)
#pragma unroll
        for (int ni = 0; ni < 2; ++ni) {
            int col = bn + wn + ni * 16 + ln;
#pragma unroll
            for (int i = 0; i < 8; ++i) {
                int row = bm + wm + mi * 16 + i + 8 * hh;
                float v = acc[mi][ni][i] * cscale;
                if (OUT_F32)
                    ((float*)Cptr)[(size_t)row * N + col] = v;
                else
                    ((unsigned short*)Cptr)[(size_t)row * N + col] = f2bf(v);
            }
        }
}

// ---------------------------------------------------------------------------
// Dual online softmax update (flash-attention style) for one 16x64 score slab.
// C/D layout: lane = column (16-lane halves), VGPR i = row {i | i+8}.
// Row stats live per VGPR index, uniform across each 16-lane half.
// ---------------------------------------------------------------------------
__device__ __forceinline__ void softmax_update(v8f (&s)[4], float (&m)[8], float (&l)[8],
                                               v8f (&o)[4], unsigned short* Prow) {
    int lane = threadIdx.x & 31;
    int ln = lane & 15, hh = lane >> 4;
#pragma unroll
    for (int i = 0; i < 8; ++i) {
        // row max across the 64 columns of this slab
        float t = fmaxf(fmaxf(s[0][i], s[1][i]), fmaxf(s[2][i], s[3][i]));
#pragma unroll
        for (int off = 8; off >= 1; off >>= 1) t = fmaxf(t, __shfl_xor(t, off, 32));
        float mn = fmaxf(m[i], t);
        float sc = __expf(m[i] - mn);
        m[i] = mn;
        l[i] *= sc;
#pragma unroll
        for (int dt = 0; dt < 4; ++dt) o[dt][i] *= sc;
        // p = exp(s - m), row-sum, stage P (bf16) for the P*V WMMA
        float part = 0.f;
#pragma unroll
        for (int nt = 0; nt < 4; ++nt) {
            float p = __expf(s[nt][i] - mn);
            part += p;
            Prow[(size_t)(i + 8 * hh) * 64 + nt * 16 + ln] = f2bf(p);
        }
#pragma unroll
        for (int off = 8; off >= 1; off >>= 1) part += __shfl_xor(part, off, 32);
        l[i] += part;
    }
}

// ---------------------------------------------------------------------------
// Streaming differential attention. grid = (NQ/64, B*H), block = 128 (4 waves).
// Each wave owns 16 query rows; KV processed in 64-row tiles. Q is pre-scaled
// by SCALE in the projection GEMM, so scores come out of WMMA ready-to-use.
// ---------------------------------------------------------------------------
__global__ __launch_bounds__(128) void diffattn_kernel(
    const unsigned short* __restrict__ Qb,   // [B*NQ, 2*DIM] bf16 (pre-scaled)
    const unsigned short* __restrict__ Kb,   // [B*NKV, 2*DIM] bf16
    const unsigned short* __restrict__ Vb,   // [B*NKV, DIM]   bf16
    const float* __restrict__ lam,           // [H]
    unsigned short* __restrict__ Ob) {       // [B*NQ, DIM]    bf16
    __shared__ unsigned short sQ1[64 * 64], sQ2[64 * 64];
    __shared__ unsigned short sK1[64 * 64], sK2[64 * 64];
    __shared__ unsigned short sV [64 * 64];
    __shared__ unsigned short sP1[64 * 64], sP2[64 * 64];   // total 56 KB LDS

    const int tid  = threadIdx.x;
    const int bh   = blockIdx.y;
    const int b    = bh >> 4, h = bh & 15;
    const int q0   = blockIdx.x * 64;
    const int wave = tid >> 5, lane = tid & 31;
    const int qr   = wave * 16;

    const size_t qks = 2 * DIM_, vs = DIM_;
    const unsigned short* Qbase  = Qb + ((size_t)b * NSEQ_ + q0) * qks + h * 2 * HD_;
    const unsigned short* Kbase0 = Kb + (size_t)b * NSEQ_ * qks + h * 2 * HD_;
    const unsigned short* Vbase0 = Vb + (size_t)b * NSEQ_ * vs  + h * HD_;

    // ---- stage Q1/Q2 (64x64 each); completion covered by the first tile's
    //      async-wait + barrier ----
#if HAVE_ASYNC_LDS
#pragma unroll
    for (int it = 0; it < 4; ++it) {
        int idx = (tid + it * 128) * 8;
        int r = idx >> 6, c = idx & 63;
        async_cp16(Qbase + (size_t)r * qks + c,       &sQ1[r * 64 + c]);
        async_cp16(Qbase + (size_t)r * qks + HD_ + c, &sQ2[r * 64 + c]);
    }
#else
#pragma unroll
    for (int it = 0; it < 8; ++it) {
        int idx = (tid + it * 128) * 4;
        int r = idx >> 6, c = idx & 63;
        *(uint2*)&sQ1[r * 64 + c] = *(const uint2*)(Qbase + (size_t)r * qks + c);
        *(uint2*)&sQ2[r * 64 + c] = *(const uint2*)(Qbase + (size_t)r * qks + HD_ + c);
    }
#endif

    v8f o1[4], o2[4];
    float m1[8], l1[8], m2[8], l2[8];
#pragma unroll
    for (int dt = 0; dt < 4; ++dt) { o1[dt] = zero8(); o2[dt] = zero8(); }
#pragma unroll
    for (int i = 0; i < 8; ++i) { m1[i] = -1e30f; m2[i] = -1e30f; l1[i] = 0.f; l2[i] = 0.f; }

    for (int k0 = 0; k0 < NSEQ_; k0 += 64) {
        __syncthreads();   // previous tile fully consumed by all waves
        const unsigned short* Kbase = Kbase0 + (size_t)k0 * qks;
        const unsigned short* Vbase = Vbase0 + (size_t)k0 * vs;
#if HAVE_ASYNC_LDS
#pragma unroll
        for (int it = 0; it < 4; ++it) {
            int idx = (tid + it * 128) * 8;
            int r = idx >> 6, c = idx & 63;
            async_cp16(Kbase + (size_t)r * qks + c,       &sK1[r * 64 + c]);
            async_cp16(Kbase + (size_t)r * qks + HD_ + c, &sK2[r * 64 + c]);
            async_cp16(Vbase + (size_t)r * vs + c,        &sV [r * 64 + c]);
        }
        async_wait0();      // s_wait_asynccnt 0 (covers Q on first iteration)
#else
#pragma unroll
        for (int it = 0; it < 8; ++it) {
            int idx = (tid + it * 128) * 4;
            int r = idx >> 6, c = idx & 63;
            *(uint2*)&sK1[r * 64 + c] = *(const uint2*)(Kbase + (size_t)r * qks + c);
            *(uint2*)&sK2[r * 64 + c] = *(const uint2*)(Kbase + (size_t)r * qks + HD_ + c);
            *(uint2*)&sV [r * 64 + c] = *(const uint2*)(Vbase + (size_t)r * vs + c);
        }
#endif
        __syncthreads();

        // S = (Q*SCALE) K^T.  K tile is [kv][d] in LDS: reading it row-major
        // per-column-lane is exactly the A-fragment pattern => B = K^T.
        v16bf aq1[2], aq2[2];
#pragma unroll
        for (int ks = 0; ks < 2; ++ks) {
            aq1[ks] = frag_rowmajor(sQ1, 64, qr, ks * 32);
            aq2[ks] = frag_rowmajor(sQ2, 64, qr, ks * 32);
        }
        v8f s1[4], s2[4];
#pragma unroll
        for (int nt = 0; nt < 4; ++nt) { s1[nt] = zero8(); s2[nt] = zero8(); }
#pragma unroll
        for (int nt = 0; nt < 4; ++nt)
#pragma unroll
            for (int ks = 0; ks < 2; ++ks) {
                v16bf bk1 = frag_rowmajor(sK1, 64, nt * 16, ks * 32);
                v16bf bk2 = frag_rowmajor(sK2, 64, nt * 16, ks * 32);
                s1[nt] = wmma_bf16(aq1[ks], bk1, s1[nt]);
                s2[nt] = wmma_bf16(aq2[ks], bk2, s2[nt]);
            }

        softmax_update(s1, m1, l1, o1, &sP1[(size_t)qr * 64]);
        softmax_update(s2, m2, l2, o2, &sP2[(size_t)qr * 64]);

        // O += P * V  (P staged through this wave's private LDS slice)
#pragma unroll
        for (int ks = 0; ks < 2; ++ks) {
            v16bf ap1 = frag_rowmajor(sP1, 64, qr, ks * 32);
            v16bf ap2 = frag_rowmajor(sP2, 64, qr, ks * 32);
#pragma unroll
            for (int dt = 0; dt < 4; ++dt) {
                v16bf bv = frag_kn(sV, 64, dt * 16, ks * 32);
                o1[dt] = wmma_bf16(ap1, bv, o1[dt]);
                o2[dt] = wmma_bf16(ap2, bv, o2[dt]);
            }
        }
    }

    // epilogue: out = O1/l1 - lam * O2/l2  (bf16 workspace for the Wo GEMM)
    const float lm = lam[h];
    const int ln = lane & 15, hh = lane >> 4;
#pragma unroll
    for (int dt = 0; dt < 4; ++dt)
#pragma unroll
        for (int i = 0; i < 8; ++i) {
            int row = q0 + qr + i + 8 * hh;
            float v = o1[dt][i] * (1.f / l1[i]) - lm * o2[dt][i] * (1.f / l2[i]);
            Ob[((size_t)b * NSEQ_ + row) * DIM_ + h * HD_ + dt * 16 + ln] = f2bf(v);
        }
}

// ---------------------------------------------------------------------------
extern "C" void kernel_launch(void* const* d_in, const int* in_sizes, int n_in,
                              void* d_out, int out_size, void* d_ws, size_t ws_size,
                              hipStream_t stream) {
    (void)in_sizes; (void)n_in; (void)out_size; (void)ws_size;
    const float* Drift = (const float*)d_in[0];
    const float* Ocean = (const float*)d_in[1];
    const float* Wq    = (const float*)d_in[2];
    const float* Wk    = (const float*)d_in[3];
    const float* Wv    = (const float*)d_in[4];
    const float* Wo    = (const float*)d_in[5];
    const float* lq1   = (const float*)d_in[6];
    const float* lk1   = (const float*)d_in[7];
    const float* lq2   = (const float*)d_in[8];
    const float* lk2   = (const float*)d_in[9];

    char* ws = (char*)d_ws;
    float* lam = (float*)ws;
    unsigned short* Qb = (unsigned short*)(ws + 256);
    unsigned short* Kb = Qb + (size_t)B_ * NSEQ_ * 2 * DIM_;   // +16 MB
    unsigned short* Vb = Kb + (size_t)B_ * NSEQ_ * 2 * DIM_;   // +16 MB
    unsigned short* Ab = Vb + (size_t)B_ * NSEQ_ * DIM_;       // +8 MB (attn out)
    float* out = (float*)d_out;

    const int M = B_ * NSEQ_;   // 4096

    lambda_kernel<<<dim3(H_), dim3(32), 0, stream>>>(lq1, lk1, lq2, lk2, lam);

    // Q projection pre-scaled by SCALE (exact: power of two)
    gemm_wmma<true, false><<<dim3(2 * DIM_ / 128, M / 128), dim3(256), 0, stream>>>(
        Drift, Wq, Qb, M, DIM_, 2 * DIM_, SCALE_);
    gemm_wmma<true, false><<<dim3(2 * DIM_ / 128, M / 128), dim3(256), 0, stream>>>(
        Ocean, Wk, Kb, M, DIM_, 2 * DIM_, 1.0f);
    gemm_wmma<true, false><<<dim3(DIM_ / 128, M / 128), dim3(256), 0, stream>>>(
        Ocean, Wv, Vb, M, DIM_, DIM_, 1.0f);

    diffattn_kernel<<<dim3(NSEQ_ / 64, B_ * H_), dim3(128), 0, stream>>>(
        Qb, Kb, Vb, lam, Ab);

    gemm_wmma<false, true><<<dim3(DIM_ / 128, M / 128), dim3(256), 0, stream>>>(
        Ab, Wo, out, M, DIM_, DIM_, 1.0f);
}
